// AFM_51831665328206
// MI455X (gfx1250) — matrix operand order repaired and checked by reading, not compiled, not run on hardware
//
#include <hip/hip_runtime.h>

// CDNA5 / gfx1250 WMMA types
typedef __attribute__((ext_vector_type(16))) _Float16 v16h;
typedef __attribute__((ext_vector_type(8)))  float    v8f;

#define N_FIELDS 40
#define K_DIM    64
#define ATTN     64
#define NPAIRS   (N_FIELDS * N_FIELDS)   // 1600
#define NTILES   (NPAIRS / 16)           // 100, exact
#define THREADS  128
#define NWAVES   4

__global__ __launch_bounds__(THREADS, 1)
void afm_wmma_kernel(const int* __restrict__ x,
                     const float* __restrict__ lin_table,
                     const float* __restrict__ V_table,
                     const float* __restrict__ W1,
                     const float* __restrict__ b1,
                     const float* __restrict__ W2,
                     const float* __restrict__ b2,
                     const float* __restrict__ Wf,
                     const float* __restrict__ bf,
                     float* __restrict__ out)
{
    __shared__ int      s_idx[N_FIELDS];
    __shared__ float    s_lin[N_FIELDS];
    __shared__ float    s_embd[N_FIELDS][K_DIM];     // 10240 B, f32 for accuracy
    __shared__ _Float16 s_w1[K_DIM * ATTN];          // 8192 B, f16 for WMMA B
    __shared__ float    s_scores[NPAIRS];            // 6400 B, scores -> probs
    __shared__ float    s_red[THREADS];              // reduction scratch

    const int b   = blockIdx.x;
    const int tid = threadIdx.x;
    const int ln  = tid & 31;      // lane in wave (wave32)
    const int wv  = tid >> 5;      // wave id 0..3
    const int lh  = ln & 15;       // lane within 16-lane half
    const int hi  = ln >> 4;       // which half (K-split for A/B fragments)

    // ---------------- Phase 0: stage tables ----------------
    if (tid < N_FIELDS) {
        const int ix = x[b * N_FIELDS + tid];
        s_idx[tid] = ix;
        s_lin[tid] = lin_table[ix];
    }
    for (int e = tid; e < K_DIM * ATTN; e += THREADS)
        s_w1[e] = (_Float16)W1[e];
    __syncthreads();

    // embedding gather, 16B vector loads (V_table rows are 256B)
    for (int e4 = tid; e4 < N_FIELDS * (K_DIM / 4); e4 += THREADS) {
        const int f  = e4 >> 4;
        const int k4 = e4 & 15;
        const float4 v = ((const float4*)(V_table + (size_t)s_idx[f] * K_DIM))[k4];
        ((float4*)&s_embd[f][0])[k4] = v;
    }
    __syncthreads();

    // ---------------- Preload B fragments of W1 (per lane) ----------------
    // B fragment k order chosen here (32x16, 16-bit; A uses the same): lane holds column N = lh; lanes 0-15 hold
    // K=0..15 (2 per VGPR), lanes 16-31 hold K=16..31.  kt selects K 0..31 / 32..63.
    v16h Bf[2][4];
    #pragma unroll
    for (int kt = 0; kt < 2; ++kt) {
        #pragma unroll
        for (int nt = 0; nt < 4; ++nt) {
            const int n     = nt * 16 + lh;
            const int kbase = kt * 32 + hi * 16;
            v16h bv;
            #pragma unroll
            for (int r = 0; r < 8; ++r) {
                bv[2 * r]     = s_w1[(kbase + 2 * r)     * ATTN + n];
                bv[2 * r + 1] = s_w1[(kbase + 2 * r + 1) * ATTN + n];
            }
            Bf[kt][nt] = bv;
        }
    }
    float w2v[4], b1v[4];
    #pragma unroll
    for (int nt = 0; nt < 4; ++nt) {
        w2v[nt] = W2[nt * 16 + lh];
        b1v[nt] = b1[nt * 16 + lh];
    }
    const float b2s = b2[0];

    // ---------------- Phase 1: pw @ W1 via WMMA, -> scores ----------------
    for (int tile = wv; tile < NTILES; tile += NWAVES) {
        // A fragment: row M = lh -> pair index; K order matches B fragments:
        // reg r holds K = hi*16 + 2*r.
        const int pair = tile * 16 + lh;
        const int fi = pair / N_FIELDS;
        const int fj = pair % N_FIELDS;
        const float* ei = &s_embd[fi][0];
        const float* ej = &s_embd[fj][0];
        v16h A[2];
        #pragma unroll
        for (int kt = 0; kt < 2; ++kt) {
            v16h av;
            #pragma unroll
            for (int r = 0; r < 8; ++r) {
                const int k = kt * 32 + hi * 16 + 2 * r;
                av[2 * r]     = (_Float16)(ei[k]     * ej[k]);
                av[2 * r + 1] = (_Float16)(ei[k + 1] * ej[k + 1]);
            }
            A[kt] = av;
        }

        v8f c[4];
        #pragma unroll
        for (int nt = 0; nt < 4; ++nt) {
            v8f z = {};
            z = __builtin_amdgcn_wmma_f32_16x16x32_f16(false, A[0], false, Bf[0][nt],
                                                       (short)0, z, false, false);
            z = __builtin_amdgcn_wmma_f32_16x16x32_f16(false, A[1], false, Bf[1][nt],
                                                       (short)0, z, false, false);
            c[nt] = z;
        }

        // score[pair] = sum_N relu(h+b1)*W2 ; C layout: lane N = lh,
        // reg r holds M = r (lanes 0-15) or r+8 (lanes 16-31).
        #pragma unroll
        for (int r = 0; r < 8; ++r) {
            float acc = 0.f;
            #pragma unroll
            for (int nt = 0; nt < 4; ++nt) {
                float hvl = c[nt][r] + b1v[nt];
                hvl = hvl > 0.f ? hvl : 0.f;
                acc += hvl * w2v[nt];
            }
            // reduce over the 16 lanes of each half (wave32: masks stay in-half)
            acc += __shfl_xor(acc, 1, 32);
            acc += __shfl_xor(acc, 2, 32);
            acc += __shfl_xor(acc, 4, 32);
            acc += __shfl_xor(acc, 8, 32);
            if (lh == 0)
                s_scores[tile * 16 + r + 8 * hi] = acc + b2s;
        }
    }
    __syncthreads();

    // ---------------- Phase 2: softmax over j per row i ----------------
    if (tid < N_FIELDS) {
        float* row = &s_scores[tid * N_FIELDS];
        float mx = -1e30f;
        for (int j = 0; j < N_FIELDS; ++j) mx = fmaxf(mx, row[j]);
        float se = 0.f;
        for (int j = 0; j < N_FIELDS; ++j) {
            const float e = __expf(row[j] - mx);
            row[j] = e;
            se += e;
        }
        const float inv = 1.f / se;
        for (int j = 0; j < N_FIELDS; ++j) row[j] *= inv;
    }
    __syncthreads();

    // ---------------- Phase 3: weighted_embd[k] = sum_i e_i[k] * (sum_j p_ij e_j[k]) --
    {
        const int k    = tid & 63;
        const int half = tid >> 6;
        float acc = 0.f;
        for (int i = half; i < N_FIELDS; i += 2) {
            const float* pr = &s_scores[i * N_FIELDS];
            float wi = 0.f;
            for (int j = 0; j < N_FIELDS; ++j) wi += pr[j] * s_embd[j][k];
            acc += s_embd[i][k] * wi;
        }
        s_red[tid] = acc;
    }
    __syncthreads();
    if (tid < 64)
        s_red[tid] = (s_red[tid] + s_red[tid + 64]) * Wf[tid];
    __syncthreads();
    if (tid == 0) {
        float inter = 0.f;
        for (int k = 0; k < 64; ++k) inter += s_red[k];
        float lin = 0.f;
        for (int f = 0; f < N_FIELDS; ++f) lin += s_lin[f];
        const float z = lin + inter + bf[0];
        out[b] = 1.f / (1.f + __expf(-z));
    }
}

extern "C" void kernel_launch(void* const* d_in, const int* in_sizes, int n_in,
                              void* d_out, int out_size, void* d_ws, size_t ws_size,
                              hipStream_t stream)
{
    const int*   x         = (const int*)d_in[0];
    const float* lin_table = (const float*)d_in[1];
    const float* V_table   = (const float*)d_in[2];
    const float* W1        = (const float*)d_in[3];
    const float* b1        = (const float*)d_in[4];
    const float* W2        = (const float*)d_in[5];
    const float* b2        = (const float*)d_in[6];
    const float* Wf        = (const float*)d_in[7];
    const float* bf        = (const float*)d_in[8];
    float*       out       = (float*)d_out;

    const int batch = out_size;  // 2048: one workgroup per batch element
    hipLaunchKernelGGL(afm_wmma_kernel, dim3(batch), dim3(THREADS), 0, stream,
                       x, lin_table, V_table, W1, b1, W2, b2, Wf, bf, out);
}
